// MultiHeadSelfAttention_987842478514
// MI455X (gfx1250) — matrix operand order repaired and checked by reading, not compiled, not run on hardware
//
#include <hip/hip_runtime.h>

// ---------------- problem constants ----------------
#define B_  2
#define S_  2048
#define D_  1024
#define H_  16
#define HD_ 64
constexpr int Mtot = B_ * S_;   // 4096 rows for the projection GEMMs
constexpr int Kdim = D_;        // 1024 reduction dim for projections
constexpr int LDSPAD = 40;      // LDS B-tile row stride in bf16 (80B; banks spread)
constexpr int SCPAD  = 2052;    // attention score row stride in floats

// ---------------- types ----------------
typedef __bf16 bf16_t;
typedef __attribute__((ext_vector_type(16))) __bf16 v16bf;
typedef __attribute__((ext_vector_type(8)))  __bf16 v8bf;
typedef __attribute__((ext_vector_type(4)))  __bf16 v4bf;
typedef __attribute__((ext_vector_type(8)))  float  v8f;
typedef __attribute__((ext_vector_type(4)))  float  v4f;
typedef __attribute__((ext_vector_type(4)))  int    v4i;

union FragU { v16bf v; v8bf h[2]; };

// D = A(16x32 bf16) x B(32x16 bf16) + C(16x16 f32)
__device__ __forceinline__ v8f wmma_bf16(v16bf a, v16bf b, v8f c) {
    return __builtin_amdgcn_wmma_f32_16x16x32_bf16(
        /*neg_a=*/false, a, /*neg_b=*/false, b,
        /*c_mod=*/(short)0, c, /*reuse_a=*/false, /*reuse_b=*/false);
}

// ---- async global -> LDS staging (gfx1250 ASYNCcnt path), with fallback ----
__device__ __forceinline__ void async_copy_b128(const bf16_t* g, bf16_t* l) {
#if __has_builtin(__builtin_amdgcn_global_load_async_to_lds_b128)
    __builtin_amdgcn_global_load_async_to_lds_b128(
        (__attribute__((address_space(1))) v4i*)g,
        (__attribute__((address_space(3))) v4i*)l, 0, 0);
#else
    *(v8bf*)l = *(const v8bf*)g;
#endif
}
__device__ __forceinline__ void async_wait_all() {
#if __has_builtin(__builtin_amdgcn_global_load_async_to_lds_b128)
#if __has_builtin(__builtin_amdgcn_s_wait_asynccnt)
    __builtin_amdgcn_s_wait_asynccnt(0);
#else
    asm volatile("s_wait_asynccnt 0x0" ::: "memory");
#endif
#endif
}
// 16 bf16 (32B) per thread
__device__ __forceinline__ void stage32B(const bf16_t* g, bf16_t* l) {
    async_copy_b128(g, l);
    async_copy_b128(g + 8, l + 8);
}

// A-matrix 16x32 bf16 fragment: lane<16 -> row=row0+lane, K = k0..k0+15
// (contiguous); lane>=16 -> K = k0+16..k0+31.
__device__ __forceinline__ v16bf load_fragA(const bf16_t* base, int ld, int lane,
                                            int row0, int k0) {
    int r  = row0 + (lane & 15);
    int kc = k0 + ((lane & 16) ? 16 : 0);
    FragU f;
    const bf16_t* p = base + (size_t)r * ld + kc;
    f.h[0] = *(const v8bf*)(p);
    f.h[1] = *(const v8bf*)(p + 8);
    return f.v;
}

// B-matrix 32x16 bf16 fragment: lane holds column n = n0 + (lane&15);
// lanes 0-15 hold K = k0..k0+15, lanes 16-31 hold K = k0+16..k0+31 (contiguous).
// "base" is row-major [n][k] (i.e. B^T), so this is one contiguous 32B load.
__device__ __forceinline__ v16bf load_fragB_rm(const bf16_t* base, int ld, int lane,
                                               int n0, int k0) {
    int r  = n0 + (lane & 15);
    int kc = k0 + ((lane & 16) ? 16 : 0);
    return *(const v16bf*)(base + (size_t)r * ld + kc);
}

// ---------------- kernel 1: fp32 -> bf16 convert ----------------
__global__ void cvt_f32_bf16(const float4* __restrict__ in,
                             bf16_t* __restrict__ out, int n4) {
    int i = blockIdx.x * blockDim.x + threadIdx.x;
    if (i < n4) {
        float4 v = in[i];
        v4bf r = { (bf16_t)v.x, (bf16_t)v.y, (bf16_t)v.z, (bf16_t)v.w };
        *(v4bf*)(out + (size_t)i * 4) = r;
    }
}

// ---------------- kernel 2: projection GEMM ----------------
// C[m,n] = sum_k A[m,k] * W[n,k] + bias[n];  M=4096, N=1024, K=1024.
// Block: 256 threads = 8 waves. Block tile 128(M) x 128(N); wave w owns rows
// [w*16, w*16+16) x 8 WMMA n-tiles (64 acc VGPRs). W k-slice (128 rows x 32 k)
// double-buffered in LDS (padded stride, bank-conflict-free) via async
// global->LDS copies: one barrier per k-step; 8 B-fragments preloaded so all
// 16 ds_load_b128 pipeline in one clause before 8 back-to-back WMMAs.
// mode 0: out bf16 [B,H,S,64]   (Q, K)
// mode 1: out bf16 [B,H,64,S]   (V transposed)
// mode 2: out f32  [B,S,D]      (final output projection)
__global__ void __launch_bounds__(256)
proj_gemm(const bf16_t* __restrict__ A,    // [Mtot, Kdim]
          const bf16_t* __restrict__ W,    // [1024, Kdim] (out_features major)
          const float*  __restrict__ bias, // [1024]
          bf16_t* __restrict__ outb,
          float*  __restrict__ outf,
          int mode) {
    __shared__ __align__(16) bf16_t Bs[2][128 * LDSPAD];
    const int tid  = threadIdx.x;
    const int lane = tid & 31;
    const int wave = tid >> 5;                  // 0..7
    const int m0   = blockIdx.x * 128 + wave * 16;
    const int n0   = blockIdx.y * 128;

    // staging: thread t covers row = t>>1 (0..127), 16 elems at ch = (t&1)*16
    const int srow = tid >> 1;
    const int sch  = (tid & 1) * 16;
    const bf16_t* wsrc = W + (size_t)(n0 + srow) * Kdim + sch;
    bf16_t* l0 = &Bs[0][srow * LDSPAD + sch];
    bf16_t* l1 = &Bs[1][srow * LDSPAD + sch];

    stage32B(wsrc, l0);                        // prefetch k-slice 0
    v8f acc[8] = {};
    for (int k = 0; k < Kdim; k += 32) {
        const int pb = (k >> 5) & 1;
        const bf16_t* cur = &Bs[pb][0];
        bf16_t* nxt = pb ? l0 : l1;
        async_wait_all();
        __syncthreads();
        if (k + 32 < Kdim) stage32B(wsrc + (k + 32), nxt);
        v16bf a = load_fragA(A, Kdim, lane, m0, k);
        const int kc = (lane & 16) ? 16 : 0;
        FragU bfr[8];
        #pragma unroll
        for (int t = 0; t < 8; ++t) {          // one big ds_load clause
            const bf16_t* p = &cur[(t * 16 + (lane & 15)) * LDSPAD + kc];
            bfr[t].h[0] = *(const v8bf*)(p);
            bfr[t].h[1] = *(const v8bf*)(p + 8);
        }
        #pragma unroll
        for (int t = 0; t < 8; ++t)            // 8 back-to-back WMMAs
            acc[t] = wmma_bf16(a, bfr[t].v, acc[t]);
    }

    // epilogue: C layout (ISA 7.12.2): VGPR i, lanes 0-15 -> M=i, N=lane;
    // lanes 16-31 -> M=i+8, N=lane-16.
    const int moff = (lane & 16) ? 8 : 0;
    #pragma unroll
    for (int t = 0; t < 8; ++t) {
        int nl = n0 + t * 16 + (lane & 15);
        float bv = bias[nl];
        int h = nl >> 6, hd = nl & 63;
        #pragma unroll
        for (int i = 0; i < 8; ++i) {
            int m = m0 + i + moff;
            int b = m >> 11, s = m & (S_ - 1);
            float val = acc[t][i] + bv;
            if (mode == 0)
                outb[(((size_t)(b * H_ + h)) * S_ + s) * HD_ + hd] = (bf16_t)val;
            else if (mode == 1)
                outb[(((size_t)(b * H_ + h)) * HD_ + hd) * S_ + s] = (bf16_t)val;
            else
                outf[(size_t)m * D_ + nl] = val;
        }
    }
}

// ---------------- kernel 3: fused attention ----------------
// One workgroup per (b,h, 16-query-row block).  256 threads = 8 waves.
// LDS: sc[16][SCPAD] f32 scores/probs (~128KB) + part[4][256] f32 partials.
__global__ void __launch_bounds__(256)
attn_fused(const bf16_t* __restrict__ Q,   // [B*H, S, 64]
           const bf16_t* __restrict__ Kk,  // [B*H, S, 64]
           const bf16_t* __restrict__ Vt,  // [B*H, 64, S]
           float* __restrict__ attn,       // [B*H, S, S]
           bf16_t* __restrict__ ctx) {     // [B, S, D] bf16
    extern __shared__ float sm[];
    float* sc   = sm;                 // 16 * SCPAD
    float* part = sm + 16 * SCPAD;    // 4 * 256

    const int tid  = threadIdx.x;
    const int lane = tid & 31;
    const int wave = tid >> 5;                 // 0..7
    const int bh   = blockIdx.x >> 7;          // 128 query-blocks per (b,h)
    const int m0   = (blockIdx.x & 127) * 16;

    const bf16_t* Qb = Q  + (size_t)bh * S_ * HD_;
    const bf16_t* Kb = Kk + (size_t)bh * S_ * HD_;
    const bf16_t* Vb = Vt + (size_t)bh * HD_ * S_;
    float* attn_b    = attn + (size_t)bh * S_ * S_;

    // ---- phase 1: scores = (Q Kt) * 1/8 into LDS ----
    v16bf q0 = load_fragA(Qb, HD_, lane, m0, 0);
    v16bf q1 = load_fragA(Qb, HD_, lane, m0, 32);
    const int moff = (lane & 16) ? 8 : 0;
    for (int t = 0; t < 16; ++t) {
        int nb = wave * 256 + t * 16;
        v16bf k0 = load_fragB_rm(Kb, HD_, lane, nb, 0);
        v16bf k1 = load_fragB_rm(Kb, HD_, lane, nb, 32);
        v8f s = {};
        s = wmma_bf16(q0, k0, s);
        s = wmma_bf16(q1, k1, s);
        int n = nb + (lane & 15);
        #pragma unroll
        for (int i = 0; i < 8; ++i)
            sc[(i + moff) * SCPAD + n] = s[i] * 0.125f;
    }
    __syncthreads();

    // ---- phase 2: softmax rows (wave handles rows 2w, 2w+1); write attn ----
    for (int rr = 0; rr < 2; ++rr) {
        int r = wave * 2 + rr;
        float* row = sc + r * SCPAD;
        float mx = -3.0e38f;
        for (int c = lane; c < S_; c += 32) mx = fmaxf(mx, row[c]);
        #pragma unroll
        for (int o = 16; o; o >>= 1) mx = fmaxf(mx, __shfl_xor(mx, o, 32));
        float sum = 0.f;
        for (int c = lane; c < S_; c += 32) {
            float e = __expf(row[c] - mx);
            row[c] = e;
            sum += e;
        }
        #pragma unroll
        for (int o = 16; o; o >>= 1) sum += __shfl_xor(sum, o, 32);
        float inv = 1.f / sum;
        float* arow = attn_b + (size_t)(m0 + r) * S_;
        for (int c = lane * 4; c < S_; c += 128) {   // b128 nontemporal stream
            v4f p = *(v4f*)(row + c) * inv;
            *(v4f*)(row + c) = p;
            __builtin_nontemporal_store(p, (v4f*)(arow + c));
        }
    }
    __syncthreads();

    // ---- phase 3: ctx = P(16x2048) @ V(2048x64) ----
    // wave -> (n-tile = wave&3, k-half = wave>>2); halves reduced via LDS.
    const int ntile = wave & 3;
    const int khalf = wave >> 2;
    const int nb = ntile * 16;
    v8f acc = {};
    for (int kk = khalf * 1024; kk < khalf * 1024 + 1024; kk += 32) {
        // A fragment from LDS probs (f32 -> bf16)
        int r  = lane & 15;
        int kc = kk + ((lane & 16) ? 16 : 0);
        const float* s0 = sc + r * SCPAD + kc;
        FragU af;
        #pragma unroll
        for (int i = 0; i < 8; ++i) {
            af.h[0][i] = (bf16_t)s0[i];
            af.h[1][i] = (bf16_t)s0[i + 8];
        }
        v16bf b = load_fragB_rm(Vb, S_, lane, nb, kk);  // Vt rows = B columns
        acc = wmma_bf16(af.v, b, acc);
    }
    if (khalf == 1) {
        float* p = part + ntile * 256;
        #pragma unroll
        for (int i = 0; i < 8; ++i) p[i * 32 + lane] = acc[i];
    }
    __syncthreads();
    if (khalf == 0) {
        const float* p = part + ntile * 256;
        int b = bh >> 4, h = bh & 15;
        int nl = nb + (lane & 15);
        #pragma unroll
        for (int i = 0; i < 8; ++i) {
            float v = acc[i] + p[i * 32 + lane];
            int s = m0 + i + moff;
            ctx[((size_t)b * S_ + s) * D_ + h * HD_ + nl] = (bf16_t)v;
        }
    }
}

// ---------------- launcher ----------------
extern "C" void kernel_launch(void* const* d_in, const int* in_sizes, int n_in,
                              void* d_out, int out_size, void* d_ws, size_t ws_size,
                              hipStream_t stream) {
    (void)in_sizes; (void)n_in; (void)out_size; (void)ws_size;
    const float* x  = (const float*)d_in[0];
    const float* Wq = (const float*)d_in[1];
    const float* bq = (const float*)d_in[2];
    const float* Wk = (const float*)d_in[3];
    const float* bk = (const float*)d_in[4];
    const float* Wv = (const float*)d_in[5];
    const float* bv = (const float*)d_in[6];
    const float* Wo = (const float*)d_in[7];
    const float* bo = (const float*)d_in[8];

    float* out  = (float*)d_out;                       // [B,S,D]
    float* attn = out + (size_t)B_ * S_ * D_;          // [B,H,S,S]

    char* ws = (char*)d_ws;
    size_t off = 0;
    auto take = [&](size_t bytes) { char* p = ws + off; off += bytes; return p; };
    const size_t xN = (size_t)Mtot * D_;   // 4.19M elems
    const size_t wN = (size_t)D_ * D_;     // 1.05M elems
    bf16_t* xb   = (bf16_t*)take(xN * 2);
    bf16_t* Wqb  = (bf16_t*)take(wN * 2);
    bf16_t* Wkb  = (bf16_t*)take(wN * 2);
    bf16_t* Wvb  = (bf16_t*)take(wN * 2);
    bf16_t* Wob  = (bf16_t*)take(wN * 2);
    bf16_t* Qb   = (bf16_t*)take(xN * 2);  // [B,H,S,64]
    bf16_t* Kb   = (bf16_t*)take(xN * 2);  // [B,H,S,64]
    bf16_t* Vtb  = (bf16_t*)take(xN * 2);  // [B,H,64,S]
    bf16_t* ctxb = (bf16_t*)take(xN * 2);  // [B,S,D]

    // 1) fp32 -> bf16 converts
    {
        int n4 = (int)(xN / 4);
        cvt_f32_bf16<<<(n4 + 255) / 256, 256, 0, stream>>>((const float4*)x, xb, n4);
        int w4 = (int)(wN / 4);
        cvt_f32_bf16<<<(w4 + 255) / 256, 256, 0, stream>>>((const float4*)Wq, Wqb, w4);
        cvt_f32_bf16<<<(w4 + 255) / 256, 256, 0, stream>>>((const float4*)Wk, Wkb, w4);
        cvt_f32_bf16<<<(w4 + 255) / 256, 256, 0, stream>>>((const float4*)Wv, Wvb, w4);
        cvt_f32_bf16<<<(w4 + 255) / 256, 256, 0, stream>>>((const float4*)Wo, Wob, w4);
    }

    // 2) Q/K/V projections (WMMA bf16)
    dim3 pgrid(Mtot / 128, D_ / 128);
    proj_gemm<<<pgrid, 256, 0, stream>>>(xb, Wqb, bq, Qb,  nullptr, 0);
    proj_gemm<<<pgrid, 256, 0, stream>>>(xb, Wkb, bk, Kb,  nullptr, 0);
    proj_gemm<<<pgrid, 256, 0, stream>>>(xb, Wvb, bv, Vtb, nullptr, 1);

    // 3) fused attention: scores + softmax + attn write + ctx
    size_t smem = (size_t)(16 * SCPAD + 4 * 256) * sizeof(float);  // ~132KB
    attn_fused<<<B_ * H_ * (S_ / 16), 256, smem, stream>>>(Qb, Kb, Vtb, attn, ctxb);

    // 4) output projection -> fp32 out
    proj_gemm<<<pgrid, 256, 0, stream>>>(ctxb, Wob, bo, nullptr, out, 2);
}
